// GATv2_49108656062978
// MI455X (gfx1250) — matrix operand order repaired and checked by reading, not compiled, not run on hardware
//
#include <hip/hip_runtime.h>
#include <hip/hip_bf16.h>

#define NN 50000
#define FF 128
#define HH 4
#define CC 32
#define HC 128
#define NEG_SLOPE 0.2f

typedef __bf16 bf16_t;
typedef bf16_t v16bf __attribute__((ext_vector_type(16)));
typedef bf16_t v8bf  __attribute__((ext_vector_type(8)));
typedef float  v8f   __attribute__((ext_vector_type(8)));
typedef unsigned u32x4 __attribute__((ext_vector_type(4)));
typedef int      i32x8 __attribute__((ext_vector_type(8)));
typedef int      i32x4 __attribute__((ext_vector_type(4)));

// ---- monotone float<->uint mapping so unsigned atomicMax == float max ----
__device__ __forceinline__ unsigned encf(float f) {
    unsigned u = __float_as_uint(f);
    return (u & 0x80000000u) ? ~u : (u | 0x80000000u);
}
__device__ __forceinline__ float decf(unsigned u) {
    return (u & 0x80000000u) ? __uint_as_float(u & 0x7fffffffu)
                             : __uint_as_float(~u);
}

// ---------------------------------------------------------------------------
// Init: out = bias (folded into scatter-add target), max/denom reset
// ---------------------------------------------------------------------------
__global__ __launch_bounds__(256) void gat_init(float* __restrict__ out,
                                                const float* __restrict__ bias,
                                                unsigned* __restrict__ nmaxU,
                                                float* __restrict__ denom) {
    int idx = blockIdx.x * blockDim.x + threadIdx.x;
    if (idx < NN * HC) out[idx] = bias[idx & (HC - 1)];
    if (idx < NN * HH) { nmaxU[idx] = 0u; denom[idx] = 0.0f; }
}

// ---------------------------------------------------------------------------
// One-shot f32 -> bf16 conversions (x: 25.6MB -> 12.8MB; W: 3x64KB -> 3x32KB)
// ---------------------------------------------------------------------------
__global__ __launch_bounds__(256) void gat_cvt_x(const float* __restrict__ x,
                                                 bf16_t* __restrict__ xb) {
    int idx = blockIdx.x * blockDim.x + threadIdx.x;
    if (idx < NN * FF) xb[idx] = (bf16_t)x[idx];
}
__global__ __launch_bounds__(256) void gat_cvt_w(const float* __restrict__ Wl,
                                                 const float* __restrict__ Wr,
                                                 const float* __restrict__ Ws,
                                                 bf16_t* __restrict__ Wb) {
    int idx = blockIdx.x * blockDim.x + threadIdx.x;      // 3*128*128
    if (idx >= 3 * HC * FF) return;
    int m = idx / (HC * FF), r = idx % (HC * FF);
    const float* W = (m == 0) ? Wl : (m == 1) ? Wr : Ws;
    Wb[idx] = (bf16_t)W[r];
}

// ---------------------------------------------------------------------------
// TDM helper: stage a 16x128 bf16 tile (4KB) from global into LDS offset 0.
// D# per ISA §8: 2D tensor, data_size=2B, tile_dim0=128, tile_dim1=16,
// tensor_dim0=128, tensor_dim1=16, dim0_stride=128, type=2, count=1.
// ---------------------------------------------------------------------------
__device__ __forceinline__ void tdm_load_tile(const bf16_t* gsrc) {
    unsigned long long ga = (unsigned long long)gsrc;
    u32x4 g0;
    g0[0] = 1u;                                   // count=1 (valid descriptor)
    g0[1] = 0u;                                   // lds_addr = 0 (only LDS alloc)
    g0[2] = (unsigned)(ga & 0xffffffffull);       // global_addr[31:0]
    g0[3] = (unsigned)((ga >> 32) & 0x1ffffffull) // global_addr[56:32]
          | (2u << 30);                           // type = 2 ("image")
    i32x8 g1;
    g1[0] = (int)(1u << 16);                      // data_size=1 -> 2 bytes
    g1[1] = (int)(128u << 16);                    // tensor_dim0[15:0]=128 @ bits 63:48
    g1[2] = (int)(16u << 16);                     // dim0 hi=0; tensor_dim1[15:0]=16
    g1[3] = (int)(128u << 16);                    // dim1 hi=0; tile_dim0=128
    g1[4] = 16;                                   // tile_dim1=16; tile_dim2=0
    g1[5] = 128;                                  // tensor_dim0_stride[31:0]=128
    g1[6] = 0;                                    // stride hi / dim1_stride lo
    g1[7] = 0;
    i32x4 gz = {0, 0, 0, 0};
#if __clang_major__ >= 23
    i32x8 ge = {0, 0, 0, 0, 0, 0, 0, 0};
    __builtin_amdgcn_tensor_load_to_lds(g0, g1, gz, gz, ge, 0);
#else
    __builtin_amdgcn_tensor_load_to_lds(g0, g1, gz, gz, 0);
#endif
}

// ---------------------------------------------------------------------------
// Fused 3-way GEMM on bf16: block = one 16-row tile; A staged via TDM to LDS;
// 8 waves x (3 transforms) col tiles; v_wmma_f32_16x16x32_bf16, f32 accum.
// ---------------------------------------------------------------------------
__global__ __launch_bounds__(256) void gat_gemm(
        const bf16_t* __restrict__ xb, const bf16_t* __restrict__ Wb,
        const float* __restrict__ bl, const float* __restrict__ br,
        const float* __restrict__ bs,
        float* __restrict__ xl, float* __restrict__ xr, float* __restrict__ xs) {
    __shared__ bf16_t lds_a[16 * FF];             // 4KB A tile, LDS offset 0

    const int rt   = blockIdx.x;                  // 16-row tile
    const int ct   = threadIdx.x >> 5;            // wave id = 16-col tile (0..7)
    const int lane = threadIdx.x & 31;
    const int half = lane >> 4;
    const int lrow = lane & 15;

    // Keep the shared array referenced so its 4KB LDS allocation (at offset 0,
    // the TDM target) is not discarded.
    void* lds_keep = (void*)lds_a;
    asm volatile("" :: "s"(lds_keep));

    // Wave 0 issues the TDM DMA for the whole block's A tile, then waits.
    if (threadIdx.x < 32) {
        tdm_load_tile(xb + (size_t)rt * 16 * FF);
        __builtin_amdgcn_s_wait_tensorcnt(0);
    }
    __syncthreads();

    float* const outs[3] = {xl, xr, xs};
    const float* const bvs[3] = {bl, br, bs};
    const int col = ct * 16 + lrow;
    // LDS byte offset of this lane's A fragment (array is at LDS offset 0)
    const unsigned abase = (unsigned)((lrow * FF + half * 8) * 2);

    v8f acc0 = {}, acc1 = {}, acc2 = {};
#pragma unroll
    for (int kk = 0; kk < 4; ++kk) {              // K = 128 = 4 x 32
        // A fragment via explicit ds_load_b128 (TDM filled LDS behind the
        // compiler's back, so read it with hand-issued DS ops + dscnt wait).
        // lanes 0-15: K {0..7,16..23}; lanes 16-31: K {8..15,24..31}
        unsigned aoff = abase + (unsigned)(kk * 64);
        u32x4 alo_u, ahi_u;
        asm volatile("ds_load_b128 %0, %2\n\t"
                     "ds_load_b128 %1, %2 offset:32\n\t"
                     "s_wait_dscnt 0x0"
                     : "=v"(alo_u), "=v"(ahi_u)
                     : "v"(aoff));
        v8bf alo = __builtin_bit_cast(v8bf, alo_u);
        v8bf ahi = __builtin_bit_cast(v8bf, ahi_u);
        v16bf av;
#pragma unroll
        for (int i = 0; i < 8; ++i) { av[i] = alo[i]; av[i + 8] = ahi[i]; }

        v8f* accs[3] = {&acc0, &acc1, &acc2};
#pragma unroll
        for (int m = 0; m < 3; ++m) {
            // B(k,n) = W[n][k]; lane = col, K {8h..8h+7, 16+8h..16+8h+7}
            const bf16_t* wp = Wb + (size_t)m * HC * FF + (size_t)col * FF
                             + kk * 32 + half * 8;
            v8bf blo = *(const v8bf*)(wp);
            v8bf bhi = *(const v8bf*)(wp + 16);
            v16bf bv;
#pragma unroll
            for (int i = 0; i < 8; ++i) { bv[i] = blo[i]; bv[i + 8] = bhi[i]; }

            *accs[m] = __builtin_amdgcn_wmma_f32_16x16x32_bf16(
                false, av, false, bv, (short)0, *accs[m], false, false);
        }
    }

    // D layout: vgpr r, lanes 0-15 -> M=r, lanes 16-31 -> M=r+8; N = lane%16
    v8f* accs[3] = {&acc0, &acc1, &acc2};
#pragma unroll
    for (int m = 0; m < 3; ++m) {
        const float bcol = bvs[m][col];
        float* out = outs[m];
#pragma unroll
        for (int r = 0; r < 8; ++r) {
            int orow = rt * 16 + r + half * 8;
            out[(size_t)orow * HC + col] = (*accs[m])[r] + bcol;
        }
    }
}

// ---------------------------------------------------------------------------
// Edge scores: one wave per edge; lane = channel; per-head shfl_xor reduce.
// ---------------------------------------------------------------------------
__global__ __launch_bounds__(256) void gat_score(
        const int* __restrict__ ei,
        const float* __restrict__ xl, const float* __restrict__ xr,
        const float* __restrict__ att,
        float* __restrict__ sc, unsigned* __restrict__ nmaxU,
        int E, int ET) {
    const int wv   = blockIdx.x * 8 + (threadIdx.x >> 5);
    const int lane = threadIdx.x & 31;
    if (wv >= ET) return;
    int src, dst;
    if (wv < E) { src = ei[wv]; dst = ei[E + wv]; }
    else        { src = dst = wv - E; }                 // self-loop

    const float* pl = xl + (size_t)src * HC;
    const float* pr = xr + (size_t)dst * HC;
#pragma unroll
    for (int h = 0; h < HH; ++h) {
        float v = pl[h * CC + lane] + pr[h * CC + lane];
        v = (v > 0.0f) ? v : NEG_SLOPE * v;
        float s = v * att[h * CC + lane];
#pragma unroll
        for (int off = 16; off; off >>= 1) s += __shfl_xor(s, off, 32);
        if (lane == 0) {
            sc[(size_t)wv * HH + h] = s;
            atomicMax(&nmaxU[dst * HH + h], encf(s));
        }
    }
}

// ---------------------------------------------------------------------------
// p = exp(score - max[dst]); denom[dst] += p   (thread per (edge, head))
// ---------------------------------------------------------------------------
__global__ __launch_bounds__(256) void gat_p(
        const int* __restrict__ ei, float* __restrict__ sc,
        const unsigned* __restrict__ nmaxU, float* __restrict__ denom,
        int E, int ET) {
    int idx = blockIdx.x * blockDim.x + threadIdx.x;
    if (idx >= ET * HH) return;
    int e = idx >> 2, h = idx & 3;
    int dst = (e < E) ? ei[E + e] : (e - E);
    float m = decf(nmaxU[dst * HH + h]);
    float p = __expf(sc[idx] - m);
    sc[idx] = p;
    atomicAdd(&denom[dst * HH + h], p);
}

// ---------------------------------------------------------------------------
// Aggregation: out[dst] += xs[src] * (p / denom[dst]); wave per edge
// ---------------------------------------------------------------------------
__global__ __launch_bounds__(256) void gat_agg(
        const int* __restrict__ ei, const float* __restrict__ xs,
        const float* __restrict__ sc, const float* __restrict__ denom,
        float* __restrict__ out, int E, int ET) {
    const int wv   = blockIdx.x * 8 + (threadIdx.x >> 5);
    const int lane = threadIdx.x & 31;
    if (wv >= ET) return;
    int src, dst;
    if (wv < E) { src = ei[wv]; dst = ei[E + wv]; }
    else        { src = dst = wv - E; }

    const float* ps = xs + (size_t)src * HC;
#pragma unroll
    for (int h = 0; h < HH; ++h) {
        float alpha = sc[(size_t)wv * HH + h] / denom[dst * HH + h];
        atomicAdd(&out[(size_t)dst * HC + h * CC + lane],
                  ps[h * CC + lane] * alpha);
    }
}

extern "C" void kernel_launch(void* const* d_in, const int* in_sizes, int n_in,
                              void* d_out, int out_size, void* d_ws, size_t ws_size,
                              hipStream_t stream) {
    const float* x    = (const float*)d_in[0];
    const int*   ei   = (const int*)  d_in[1];
    const float* Wl   = (const float*)d_in[2];
    const float* bl   = (const float*)d_in[3];
    const float* Wr   = (const float*)d_in[4];
    const float* br   = (const float*)d_in[5];
    const float* Ws   = (const float*)d_in[6];
    const float* bs   = (const float*)d_in[7];
    const float* att  = (const float*)d_in[8];
    const float* bias = (const float*)d_in[9];
    float* out = (float*)d_out;

    const int E  = in_sizes[1] / 2;
    const int ET = E + NN;

    // workspace layout (~105 MB)
    float*    xl    = (float*)d_ws;
    float*    xr    = xl + (size_t)NN * HC;
    float*    xs    = xr + (size_t)NN * HC;
    float*    sc    = xs + (size_t)NN * HC;
    unsigned* nmaxU = (unsigned*)(sc + (size_t)ET * HH);
    float*    denom = (float*)(nmaxU + (size_t)NN * HH);
    bf16_t*   xb    = (bf16_t*)(denom + (size_t)NN * HH);
    bf16_t*   Wb    = xb + (size_t)NN * FF;

    gat_init<<<(NN * HC + 255) / 256, 256, 0, stream>>>(out, bias, nmaxU, denom);
    gat_cvt_x<<<(NN * FF + 255) / 256, 256, 0, stream>>>(x, xb);
    gat_cvt_w<<<(3 * HC * FF + 255) / 256, 256, 0, stream>>>(Wl, Wr, Ws, Wb);
    gat_gemm<<<NN / 16, 256, 0, stream>>>(xb, Wb, bl, br, bs, xl, xr, xs);

    const int eb = (ET + 7) / 8;
    gat_score<<<eb, 256, 0, stream>>>(ei, xl, xr, att, sc, nmaxU, E, ET);
    gat_p<<<(ET * HH + 255) / 256, 256, 0, stream>>>(ei, sc, nmaxU, denom, E, ET);
    gat_agg<<<eb, 256, 0, stream>>>(ei, xs, sc, denom, out, E, ET);
}
